// TimeLSTM_29231547416772
// MI455X (gfx1250) — matrix operand order repaired and checked by reading, not compiled, not run on hardware
//
#include <hip/hip_runtime.h>
#include <cstdint>
#include <cstring>

// ---------------------------------------------------------------------------
// TimeLSTM fused recurrent kernel for gfx1250 (MI455X).
// bs=2048, seq=365, INPUT_DIM=32, HS=64, XX_CH=5.
// One workgroup = 16 batch rows, 4 waves (128 threads).
// Wave w owns hidden columns [16w, 16w+16) of gates i / c_app / o.
// Hot loop per wave per step: 9x v_wmma_f32_16x16x32_bf16,
// 4x global_load_b128 (x tile), 4x ds_load_b128 (h tile),
// v_cvt_pk_bf16_f32 conversions, WGP-scope global_prefetch_b8.
// ---------------------------------------------------------------------------

#define BSZ    2048
#define SEQ    365
#define IND    32
#define HS     64
#define XXC    5
#define HPAD   72   // LDS row stride for h (bf16 elems); 144B keeps 16B align

typedef __bf16 v16bf __attribute__((ext_vector_type(16)));
typedef __bf16 v8bf  __attribute__((ext_vector_type(8)));
typedef float  v8f   __attribute__((ext_vector_type(8)));
typedef float  v4f   __attribute__((ext_vector_type(4)));

static __device__ __forceinline__ float fsigmoid(float x) {
    // 1 / (1 + 2^(-x*log2e))  -> v_exp_f32 + v_rcp_f32
    float e = __builtin_amdgcn_exp2f(-1.44269504088896340736f * x);
    return __builtin_amdgcn_rcpf(1.0f + e);
}

static __device__ __forceinline__ float ftanh(float x) {
    return 2.0f * fsigmoid(2.0f * x) - 1.0f;
}

static __device__ __forceinline__ v16bf cat8(v8bf lo, v8bf hi) {
    return __builtin_shufflevector(lo, hi, 0,1,2,3,4,5,6,7,8,9,10,11,12,13,14,15);
}

__global__ __launch_bounds__(128)
void timelstm_fused_kernel(const float* __restrict__ X,     // [BSZ, SEQ, IND]
                           const float* __restrict__ XX,    // [BSZ, SEQ, XXC]
                           const float* __restrict__ TD,    // [BSZ, SEQ, 1]
                           const float* __restrict__ Wx,    // [IND, 3*HS]
                           const float* __restrict__ Wxm,   // [XXC, 2*HS]
                           const float* __restrict__ Wh,    // [HS, 3*HS]
                           const float* __restrict__ Wt1,   // [1, HS]
                           const float* __restrict__ Wt,    // [1, 2*HS]
                           const float* __restrict__ Bias,  // [5*HS]
                           float* __restrict__ outSeq,      // [BSZ, SEQ, HS]
                           float* __restrict__ outH,        // [BSZ, HS]
                           float* __restrict__ outC)        // [BSZ, HS]
{
    __shared__ __align__(16) __bf16 hsh[16 * HPAD];   // h state, bf16, padded rows
    __shared__ float xxsh[16 * XXC];
    __shared__ float tdsh[16];

    const int tid     = threadIdx.x;
    const int lane    = tid & 31;
    const int wave    = tid >> 5;          // 0..3
    const int grp     = lane >> 4;         // lane group (0: lanes 0-15, 1: lanes 16-31)
    const int ln      = lane & 15;
    const int n0      = wave * 16;         // column base within each 64-wide gate
    const int n       = n0 + ln;           // hidden index 0..63 owned by this lane
    const int rowBase = blockIdx.x * 16;   // batch-row base of this workgroup

    // ---- loop-invariant per-lane scalars -------------------------------
    float wxm1[XXC], wxm2[XXC];
#pragma unroll
    for (int k = 0; k < XXC; ++k) {
        wxm1[k] = Wxm[k * (2 * HS) + n];
        wxm2[k] = Wxm[k * (2 * HS) + HS + n];
    }
    const float wt1n  = fminf(Wt1[n], 0.0f);   // torch clamps max=0
    const float wtn   = Wt[n];
    const float wt64n = Wt[HS + n];
    const float b_i   = Bias[n];
    const float b_t1  = Bias[HS + n];
    const float b_t2  = Bias[2 * HS + n];
    const float b_c   = Bias[3 * HS + n];
    const float b_o   = Bias[4 * HS + n];

    // ---- loop-invariant B fragments (bf16) -----------------------------
    // B layout (32x16, K x N): N=lane&15, elem e -> K = (e<8?0:16) + grp*8 + (e&7).
    v16bf Bx[3];        // x @ Wx,  K = 0..31
    v16bf Bh[3][2];     // h @ Wh,  K-blocks 0..31 / 32..63
#pragma unroll
    for (int g = 0; g < 3; ++g) {
        const int cb = g * HS + n0 + ln;   // absolute column in the 192-wide matrices
#pragma unroll
        for (int e = 0; e < 16; ++e) {
            const int kx = ((e >> 3) << 4) + grp * 8 + (e & 7);
            Bx[g][e] = (__bf16)Wx[kx * (3 * HS) + cb];
        }
#pragma unroll
        for (int kb = 0; kb < 2; ++kb) {
#pragma unroll
            for (int e = 0; e < 16; ++e) {
                const int kh = kb * 32 + ((e >> 3) << 4) + grp * 8 + (e & 7);
                Bh[g][kb][e] = (__bf16)Wh[kh * (3 * HS) + cb];
            }
        }
    }

    // ---- state init ----------------------------------------------------
    float c[8];
#pragma unroll
    for (int r = 0; r < 8; ++r) c[r] = 0.0f;
    for (int i = tid; i < 16 * HPAD; i += blockDim.x) hsh[i] = (__bf16)0.0f;

    const int m_row = lane & 15;                    // A-matrix row for this lane
    const long xrowBase = (long)(rowBase + m_row) * SEQ;

    for (int t = 0; t < SEQ; ++t) {
        // stage xx / td tiles for this step
        if (tid < 16 * XXC) {
            const int r = tid / XXC, k = tid % XXC;
            xxsh[tid] = XX[((long)(rowBase + r) * SEQ + t) * XXC + k];
        } else if (tid < 16 * XXC + 16) {
            const int r = tid - 16 * XXC;
            tdsh[r] = TD[(long)(rowBase + r) * SEQ + t];
        }

        // A fragment of x (16x32 bf16): elem e -> K = (e<8?0:16) + grp*8 + (e&7)
        // Lane's K set = two contiguous 8-float runs -> 4x global_load_b128.
        v16bf ax;
        {
            const float* xp = X + (xrowBase + t) * IND + grp * 8;
            const v4f x0 = *reinterpret_cast<const v4f*>(xp);
            const v4f x1 = *reinterpret_cast<const v4f*>(xp + 4);
            const v4f x2 = *reinterpret_cast<const v4f*>(xp + 16);
            const v4f x3 = *reinterpret_cast<const v4f*>(xp + 20);
#pragma unroll
            for (int e = 0; e < 4; ++e) {
                ax[e]      = (__bf16)x0[e];
                ax[e + 4]  = (__bf16)x1[e];
                ax[e + 8]  = (__bf16)x2[e];
                ax[e + 12] = (__bf16)x3[e];
            }
            if (t + 1 < SEQ)  // pull next step's x tile into the WGP-level cache
                __builtin_prefetch(X + (xrowBase + t + 1) * IND, 0, 3);
        }

        __syncthreads();   // h(t), xx(t), td(t) now valid in LDS

        // A fragments of h from LDS: two ds_load_b128 per K-block
        const int k0   = grp * 8;
        const v8bf hlo0 = *reinterpret_cast<const v8bf*>(&hsh[m_row * HPAD + k0]);
        const v8bf hhi0 = *reinterpret_cast<const v8bf*>(&hsh[m_row * HPAD + k0 + 16]);
        const v8bf hlo1 = *reinterpret_cast<const v8bf*>(&hsh[m_row * HPAD + 32 + k0]);
        const v8bf hhi1 = *reinterpret_cast<const v8bf*>(&hsh[m_row * HPAD + 32 + k0 + 16]);
        const v16bf ah0 = cat8(hlo0, hhi0);
        const v16bf ah1 = cat8(hlo1, hhi1);

        // 9 WMMAs: acc[g] = x@Wx_g + h@Wh_g
        v8f acc[3];
#pragma unroll
        for (int g = 0; g < 3; ++g) {
            v8f zc = {};
            zc = __builtin_amdgcn_wmma_f32_16x16x32_bf16(
                     false, ax,  false, Bx[g],    (short)0, zc, false, false);
            zc = __builtin_amdgcn_wmma_f32_16x16x32_bf16(
                     false, ah0, false, Bh[g][0], (short)0, zc, false, false);
            zc = __builtin_amdgcn_wmma_f32_16x16x32_bf16(
                     false, ah1, false, Bh[g][1], (short)0, zc, false, false);
            acc[g] = zc;
        }

        // elementwise gate math; D layout: reg r -> (m = r + grp*8, col n)
        float hn[8];
        const bool last = (t == SEQ - 1);
#pragma unroll
        for (int r = 0; r < 8; ++r) {
            const int m = r + grp * 8;
            const float tdv = tdsh[m];
            float d1 = 0.0f, d2 = 0.0f;
#pragma unroll
            for (int k = 0; k < XXC; ++k) {
                const float xv = xxsh[m * XXC + k];
                d1 += xv * wxm1[k];
                d2 += xv * wxm2[k];
            }
            const float i_g  = fsigmoid(acc[0][r] + b_i);
            const float t1_g = fsigmoid(d1 + ftanh(tdv * wt1n) + b_t1);
            const float t2_g = fsigmoid(d2 + ftanh(tdv * wtn)  + b_t2);
            const float capp = ftanh(acc[1][r] + b_c);
            const float it1  = i_g * t1_g;
            const float ctil = fsigmoid((1.0f - it1) * c[r] + it1 * capp);
            const float cnew = fsigmoid((1.0f - i_g) * c[r] + i_g * t2_g * capp);
            const float o_g  = fsigmoid(acc[2][r] + tdv * wt64n + b_o);
            const float hnew = o_g + ftanh(ctil);
            c[r]  = cnew;
            hn[r] = hnew;

            const long row = rowBase + m;
            outSeq[(row * SEQ + t) * HS + n] = hnew;
            if (last) {
                outH[row * HS + n] = hnew;
                outC[row * HS + n] = cnew;
            }
        }

        __syncthreads();   // all reads of hsh/xxsh done; safe to overwrite

        // publish h(t+1) as bf16 for all waves of this batch tile
#pragma unroll
        for (int r = 0; r < 8; ++r) {
            const int m = r + grp * 8;
            hsh[m * HPAD + n] = (__bf16)hn[r];
        }
    }
}

extern "C" void kernel_launch(void* const* d_in, const int* in_sizes, int n_in,
                              void* d_out, int out_size, void* d_ws, size_t ws_size,
                              hipStream_t stream) {
    (void)in_sizes; (void)n_in; (void)out_size; (void)d_ws; (void)ws_size;

    const float* X    = (const float*)d_in[0];   // x_for_h
    const float* XX   = (const float*)d_in[1];   // x_for_x
    const float* TD   = (const float*)d_in[2];   // TimeDiff
    const float* Wx   = (const float*)d_in[3];   // weights_x
    const float* Wxm  = (const float*)d_in[4];   // weights_x_maintained
    const float* Wh   = (const float*)d_in[5];   // weights_h
    const float* Wt1  = (const float*)d_in[6];   // weights_t1
    const float* Wt   = (const float*)d_in[7];   // weights_t
    const float* Bias = (const float*)d_in[8];   // bias

    float* outSeq = (float*)d_out;                            // [BSZ, SEQ, HS]
    float* outH   = outSeq + (long)BSZ * SEQ * HS;            // [BSZ, HS]
    float* outC   = outH   + (long)BSZ * HS;                  // [BSZ, HS]

    dim3 grid(BSZ / 16);   // 128 workgroups, one per 16 batch rows
    dim3 block(128);       // 4 waves; wave w owns 16 hidden columns
    timelstm_fused_kernel<<<grid, block, 0, stream>>>(
        X, XX, TD, Wx, Wxm, Wh, Wt1, Wt, Bias, outSeq, outH, outC);
}
